// NBFM_71107478552629
// MI455X (gfx1250) — matrix operand order repaired and checked by reading, not compiled, not run on hardware
//
#include <hip/hip_runtime.h>
#include <hip/hip_bf16.h>

typedef __attribute__((ext_vector_type(16))) _Float16 v16h;
typedef __attribute__((ext_vector_type(8)))  _Float16 v8h;
typedef __attribute__((ext_vector_type(8)))  float    v8f;
typedef __attribute__((ext_vector_type(4)))  float    v4f;

#define HH 128
#define WW 128
#define HWSZ 16384          // 128*128
#define CDIM 64
#define DL 64               // downsampled grid (stride 2)
#define RL 70               // ref grid with pad 7
#define NBN 49              // 7x7 neighbors
#define DVEC 576            // 64*9 patch vector

// ---------------------------------------------------------------------------
// per-(b,c) sum and sum-of-squares over HxW
__global__ void nbfm_stats(const float* __restrict__ x, float* __restrict__ sums,
                           float* __restrict__ sqs) {
    int bc = blockIdx.x;
    const float* p = x + (size_t)bc * HWSZ;
    float s = 0.f, q = 0.f;
    for (int i = threadIdx.x; i < HWSZ; i += 256) { float v = p[i]; s += v; q += v * v; }
    __shared__ float ss[256], sq[256];
    int tid = threadIdx.x;
    ss[tid] = s; sq[tid] = q; __syncthreads();
    for (int off = 128; off > 0; off >>= 1) {
        if (tid < off) { ss[tid] += ss[tid + off]; sq[tid] += sq[tid + off]; }
        __syncthreads();
    }
    if (tid == 0) { sums[bc] = ss[0]; sqs[bc] = sq[0]; }
}

// ---------------------------------------------------------------------------
// pack concat(a,b) [NCHW f32, 64ch each] -> NHWC f16 (128ch)
__global__ void nbfm_pack_concat(const float* __restrict__ a, const float* __restrict__ b,
                                 _Float16* __restrict__ out) {
    size_t idx = (size_t)blockIdx.x * 256 + threadIdx.x;
    int c = (int)(idx & 127);
    size_t pix = idx >> 7;                    // b*HWSZ + hw
    int bb = (int)(pix >> 14);
    int hw = (int)(pix & (HWSZ - 1));
    float v = (c < 64) ? a[((size_t)(bb * 64 + c)) * HWSZ + hw]
                       : b[((size_t)(bb * 64 + (c - 64))) * HWSZ + hw];
    out[idx] = (_Float16)v;
}

// ---------------------------------------------------------------------------
// pre-swizzle conv weights (OIHW f32, O=64) into WMMA B-fragment order:
// packed[((kt*4 + nt)*32 + lane)*16 + e], N=lane%16, h=lane/16: K = e<8 ? 8h+e : 16+8h+(e-8)
// K global = kt*32 + klocal, kt = p*(Cin/32)+ck, channel c = ck*32+klocal (c fastest within p)
__global__ void nbfm_pack_w(const float* __restrict__ w, _Float16* __restrict__ wpk,
                            int Cin, int k3) {
    int idx = blockIdx.x * 256 + threadIdx.x;
    int ckn = Cin >> 5;
    int KT = (k3 ? 9 : 1) * ckn;
    int total = KT * 4 * 32 * 16;
    if (idx >= total) return;
    int e    = idx & 15;
    int lane = (idx >> 4) & 31;
    int nt   = (idx >> 9) & 3;
    int kt   = idx >> 11;
    int n = nt * 16 + (lane & 15);
    int klocal = ((lane & 16) ? 8 : 0) + (e & 7) + ((e & 8) ? 16 : 0);
    int p = kt / ckn;
    int c = (kt % ckn) * 32 + klocal;
    float wv = k3 ? w[(size_t)(n * Cin + c) * 9 + p] : w[(size_t)n * Cin + c];
    wpk[idx] = (_Float16)wv;
}

// ---------------------------------------------------------------------------
// implicit-GEMM conv, pad=1 3x3 (K3=1) or 1x1 pad=0 (K3=0), Cout=64.
// one wave: 16 pixels (one row segment) x all 64 out channels (4 N-tiles).
// A fragment loaded once, reused by 4 back-to-back WMMAs; K-loop fully unrolled.
template <int CIN, int K3>
__global__ void nbfm_conv_wmma(const _Float16* __restrict__ act,
                               const _Float16* __restrict__ wpk,
                               const float* __restrict__ bias,
                               float* __restrict__ outf32,       // NCHW, may be null
                               _Float16* __restrict__ outf16,    // NHWC, may be null
                               int dorelu) {
    constexpr int CKN = CIN / 32;
    constexpr int NP  = K3 ? 9 : 1;
    int wave_id = blockIdx.x * 8 + (threadIdx.x >> 5);
    int lane = threadIdx.x & 31;
    int tile = wave_id & 1023;          // HWSZ/16 tiles per batch image
    int b    = wave_id >> 10;
    int m0 = tile << 4;
    int y  = m0 >> 7;
    int x0 = m0 & 127;
    int chalf = (lane & 16) ? 8 : 0;
    int lx = lane & 15;

    v8f acc[4];
#pragma unroll
    for (int nt = 0; nt < 4; ++nt) acc[nt] = (v8f){0.f, 0.f, 0.f, 0.f, 0.f, 0.f, 0.f, 0.f};

#pragma unroll
    for (int p = 0; p < NP; ++p) {
        const int dy = K3 ? (p / 3 - 1) : 0;
        const int dx = K3 ? (p % 3 - 1) : 0;
        int yy = y + dy;
        int xx = x0 + lx + dx;
        bool valid = (yy >= 0) && (yy < HH) && (xx >= 0) && (xx < WW);
        const _Float16* ap = act + ((size_t)(b * HWSZ + yy * WW + xx)) * CIN;
        // prefetch next position's weight fragments (4KB per (p,ck) step)
        if (p + 1 < NP)
            __builtin_prefetch(wpk + ((size_t)((p + 1) * CKN) * 4 * 32 * 16), 0, 1);
#pragma unroll
        for (int ck = 0; ck < CKN; ++ck) {
            int c0 = (ck << 5) + chalf;
            v16h a;
            if (valid) {
                v8h lo = *(const v8h*)(ap + c0);
                v8h hi = *(const v8h*)(ap + c0 + 16);
#pragma unroll
                for (int i = 0; i < 8; ++i) { a[i] = lo[i]; a[i + 8] = hi[i]; }
            } else {
#pragma unroll
                for (int i = 0; i < 16; ++i) a[i] = (_Float16)0.f;
            }
            const _Float16* wp = wpk + (((size_t)((p * CKN + ck) * 4) * 32) << 4);
#pragma unroll
            for (int nt = 0; nt < 4; ++nt) {
                v16h bf = *(const v16h*)(wp + (((size_t)nt * 32 + lane) << 4));
                acc[nt] = __builtin_amdgcn_wmma_f32_16x16x32_f16(
                    false, a, false, bf, (short)0, acc[nt], false, false);
            }
        }
    }

    int mbase = (lane & 16) ? 8 : 0;
#pragma unroll
    for (int nt = 0; nt < 4; ++nt) {
        int n = (nt << 4) + lx;
        float bv = bias[n];
        float res[8];
#pragma unroll
        for (int j = 0; j < 8; ++j) {
            float v = acc[nt][j] + bv;
            if (dorelu) v = fmaxf(v, 0.f);
            res[j] = v;
        }
        if (outf32) {
            float* op = outf32 + ((size_t)(b * 64 + n)) * HWSZ + y * WW + x0 + mbase;
            *(v4f*)(op)     = (v4f){res[0], res[1], res[2], res[3]};
            *(v4f*)(op + 4) = (v4f){res[4], res[5], res[6], res[7]};
        }
        if (outf16) {
#pragma unroll
            for (int j = 0; j < 8; ++j)
                outf16[((size_t)(b * HWSZ + y * WW + x0 + mbase + j)) * 64 + n] =
                    (_Float16)res[j];
        }
    }
}

// ---------------------------------------------------------------------------
// out = instnorm(x; sumX,sqX) * (gamma + sigma_d) + (beta + mu_d)
__global__ void nbfm_style_combine(const float* __restrict__ x,
                                   const float* __restrict__ gamma,
                                   const float* __restrict__ beta,
                                   const float* __restrict__ sumX, const float* __restrict__ sqX,
                                   const float* __restrict__ sumD, const float* __restrict__ sqD,
                                   float* __restrict__ out) {
    size_t idx = (size_t)blockIdx.x * 256 + threadIdx.x;
    int bc = (int)(idx >> 14);
    const float N = (float)HWSZ;
    float mx = sumX[bc] / N;
    float vx = sqX[bc] / N - mx * mx;
    float rstd = rsqrtf(vx + 1e-5f);
    float md = sumD[bc] / N;
    float v1 = (sqD[bc] - N * md * md) / (N - 1.f);
    float sig = sqrtf(fmaxf(v1, 0.f));
    float xn = (x[idx] - mx) * rstd;
    out[idx] = xn * (gamma[idx] + sig) + (beta[idx] + md);
}

// ---------------------------------------------------------------------------
// unfold (3x3, stride 2, pad) + l2norm over the 576-vector; out [b][loc][576]
// internal d ordering: d = p*64 + c  (consistent across dp/rp/matched/fold)
__global__ void nbfm_unfold_norm(const float* __restrict__ x, float* __restrict__ outp,
                                 int total_locs, int pad, int dl) {
    int wid = blockIdx.x * 8 + (threadIdx.x >> 5);
    if (wid >= total_locs) return;
    int lane = threadIdx.x & 31;
    int b = wid / (dl * dl);
    int loc = wid % (dl * dl);
    int dyv = loc / dl, dxv = loc % dl;
    float vals[18];
    float sq = 0.f;
#pragma unroll
    for (int t = 0; t < 18; ++t) {
        int d = lane + 32 * t;
        int p = d >> 6, c = d & 63;
        int kh = p / 3, kw = p % 3;
        int yy = dyv * 2 + kh - pad;
        int xx = dxv * 2 + kw - pad;
        float v = 0.f;
        if (yy >= 0 && yy < HH && xx >= 0 && xx < WW)
            v = x[((size_t)(b * 64 + c)) * HWSZ + yy * WW + xx];
        vals[t] = v;
        sq += v * v;
    }
    for (int m = 16; m >= 1; m >>= 1) sq += __shfl_xor(sq, m, 32);
    float inv = 1.f / fmaxf(sqrtf(sq), 1e-12f);
    float* op = outp + (size_t)wid * DVEC;
#pragma unroll
    for (int t = 0; t < 18; ++t) op[lane + 32 * t] = vals[t] * inv;
}

// ---------------------------------------------------------------------------
// per location: 49 neighbor dots, *nbf_weight, softmax, weighted neighbor sum.
__global__ void nbfm_attn_matched(const float* __restrict__ dpn,
                                  const float* __restrict__ rpn,
                                  const float* __restrict__ nbw,
                                  float* __restrict__ mcols) {
    int bl = blockIdx.x;                 // b*4096 + l
    int b  = bl >> 12;
    int l  = bl & 4095;
    int dyv = l >> 6, dxv = l & 63;
    __shared__ float sdp[DVEC];
    __shared__ float sat[64];
    int tid = threadIdx.x;
    for (int d = tid; d < DVEC; d += 256) sdp[d] = dpn[(size_t)bl * DVEC + d];
    __syncthreads();
    int wv = tid >> 5, lane = tid & 31;
    const float* rb = rpn + (size_t)b * (RL * RL) * DVEC;
    for (int n = wv; n < NBN; n += 8) {
        int ny = n / 7, nx = n % 7;
        const float* rr = rb + (size_t)((dxv + nx) + (dyv + ny) * RL) * DVEC;
        float acc = 0.f;
#pragma unroll
        for (int t = 0; t < 18; ++t) { int d = lane + 32 * t; acc += sdp[d] * rr[d]; }
        for (int m = 16; m >= 1; m >>= 1) acc += __shfl_xor(acc, m, 32);
        if (lane == 0) sat[n] = acc * nbw[n];
    }
    __syncthreads();
    if (tid == 0) {
        float mx = sat[0];
        for (int n = 1; n < NBN; ++n) mx = fmaxf(mx, sat[n]);
        float s = 0.f;
        for (int n = 0; n < NBN; ++n) { float e = __expf(sat[n] - mx); sat[n] = e; s += e; }
        float inv = 1.f / s;
        for (int n = 0; n < NBN; ++n) sat[n] *= inv;
    }
    __syncthreads();
    for (int d = tid; d < DVEC; d += 256) {
        float acc = 0.f;
        for (int n = 0; n < NBN; ++n) {
            int ny = n / 7, nx = n % 7;
            acc += sat[n] * rb[(size_t)((dxv + nx) + (dyv + ny) * RL) * DVEC + d];
        }
        mcols[(size_t)bl * DVEC + d] = acc;
    }
}

// ---------------------------------------------------------------------------
// fold (3x3, stride 2, pad 1): scatter-add matched cols back to NCHW image
__global__ void nbfm_fold(const float* __restrict__ mcols, float* __restrict__ out) {
    size_t idx = (size_t)blockIdx.x * 256 + threadIdx.x;
    int b = (int)(idx >> 20);
    int rem = (int)(idx & ((1 << 20) - 1));
    int c = rem >> 14;
    int hw = rem & (HWSZ - 1);
    int y = hw >> 7, x = hw & 127;
    float acc = 0.f;
#pragma unroll
    for (int kh = 0; kh < 3; ++kh)
#pragma unroll
        for (int kw = 0; kw < 3; ++kw) {
            int ty = y - kh + 1, tx = x - kw + 1;
            if (ty >= 0 && tx >= 0 && !(ty & 1) && !(tx & 1)) {
                int dyv = ty >> 1, dxv = tx >> 1;
                if (dyv < DL && dxv < DL) {
                    int l = dyv * DL + dxv;
                    int p = kh * 3 + kw;
                    acc += mcols[((size_t)(b * 4096 + l)) * DVEC + p * 64 + c];
                }
            }
        }
    out[idx] = acc;
}

// ---------------------------------------------------------------------------
// depthwise 3x3 pad 1 + bias + relu (final output)
__global__ void nbfm_dwconv(const float* __restrict__ x, const float* __restrict__ w,
                            const float* __restrict__ bias, float* __restrict__ out) {
    size_t idx = (size_t)blockIdx.x * 256 + threadIdx.x;
    int b = (int)(idx >> 20);
    int rem = (int)(idx & ((1 << 20) - 1));
    int c = rem >> 14;
    int hw = rem & (HWSZ - 1);
    int y = hw >> 7, xx = hw & 127;
    float acc = bias[c];
    const float* xp = x + ((size_t)(b * 64 + c)) * HWSZ;
#pragma unroll
    for (int kh = 0; kh < 3; ++kh)
#pragma unroll
        for (int kw = 0; kw < 3; ++kw) {
            int yy = y + kh - 1, x2 = xx + kw - 1;
            if (yy >= 0 && yy < HH && x2 >= 0 && x2 < WW)
                acc += xp[yy * WW + x2] * w[c * 9 + kh * 3 + kw];
        }
    out[idx] = fmaxf(acc, 0.f);
}

// ---------------------------------------------------------------------------
extern "C" void kernel_launch(void* const* d_in, const int* in_sizes, int n_in,
                              void* d_out, int out_size, void* d_ws, size_t ws_size,
                              hipStream_t stream) {
    const float* degraded = (const float*)d_in[0];
    const float* ref      = (const float*)d_in[1];
    const float* sbw = (const float*)d_in[2];
    const float* sbb = (const float*)d_in[3];
    const float* gbw = (const float*)d_in[4];
    const float* gbb = (const float*)d_in[5];
    const float* bbw = (const float*)d_in[6];
    const float* bbb = (const float*)d_in[7];
    const float* nbw = (const float*)d_in[8];
    const float* saw = (const float*)d_in[9];
    const float* sab = (const float*)d_in[10];
    const float* gaw = (const float*)d_in[11];
    const float* gab = (const float*)d_in[12];
    const float* baw = (const float*)d_in[13];
    const float* bab = (const float*)d_in[14];
    const float* f1w = (const float*)d_in[15];
    const float* f1b = (const float*)d_in[16];
    const float* f2w = (const float*)d_in[17];
    const float* f2b = (const float*)d_in[18];

    const int BS = in_sizes[0] / (64 * HWSZ);

    char* ws = (char*)d_ws;
    size_t off = 0;
    auto take = [&](size_t bytes) -> char* {
        char* p = ws + off;
        off += (bytes + 255) & ~(size_t)255;
        return p;
    };

    float* sumD = (float*)take((size_t)BS * 64 * 4);
    float* sqD  = (float*)take((size_t)BS * 64 * 4);
    float* sumR = (float*)take((size_t)BS * 64 * 4);
    float* sqR  = (float*)take((size_t)BS * 64 * 4);
    float* sumM = (float*)take((size_t)BS * 64 * 4);
    float* sqM  = (float*)take((size_t)BS * 64 * 4);

    _Float16* wpk_sb = (_Float16*)take((size_t)36 * 2048 * 2);
    _Float16* wpk_sa = (_Float16*)take((size_t)36 * 2048 * 2);
    _Float16* wpk_gb = (_Float16*)take((size_t)18 * 2048 * 2);
    _Float16* wpk_ga = (_Float16*)take((size_t)18 * 2048 * 2);
    _Float16* wpk_bb = (_Float16*)take((size_t)18 * 2048 * 2);
    _Float16* wpk_ba = (_Float16*)take((size_t)18 * 2048 * 2);
    _Float16* wpk_f1 = (_Float16*)take((size_t)4 * 2048 * 2);

    _Float16* actcat  = (_Float16*)take((size_t)BS * HWSZ * 128 * 2);
    _Float16* stylef  = (_Float16*)take((size_t)BS * HWSZ * 64 * 2);
    float* gbuf = (float*)take((size_t)BS * 64 * HWSZ * 4);
    float* bbuf = (float*)take((size_t)BS * 64 * HWSZ * 4);
    float* ref2 = (float*)take((size_t)BS * 64 * HWSZ * 4);   // later reused as matched
    float* dpn  = (float*)take((size_t)BS * 4096 * DVEC * 4); // later reused as mstyled
    float* rpn  = (float*)take((size_t)BS * RL * RL * DVEC * 4); // later reused as fus1out
    float* mcol = (float*)take((size_t)BS * 4096 * DVEC * 4);

    float* matched = ref2;       // alias: ref2 dead after rpn is built
    float* mstyled = dpn;        // alias: dpn dead after attention
    float* fus1out = rpn;        // alias: rpn dead after attention

    dim3 T(256);
    // stats of degraded and ref
    nbfm_stats<<<dim3(BS * 64), T, 0, stream>>>(degraded, sumD, sqD);
    nbfm_stats<<<dim3(BS * 64), T, 0, stream>>>(ref, sumR, sqR);
    // weight packing
    nbfm_pack_w<<<dim3(288), T, 0, stream>>>(sbw, wpk_sb, 128, 1);
    nbfm_pack_w<<<dim3(288), T, 0, stream>>>(saw, wpk_sa, 128, 1);
    nbfm_pack_w<<<dim3(144), T, 0, stream>>>(gbw, wpk_gb, 64, 1);
    nbfm_pack_w<<<dim3(144), T, 0, stream>>>(gaw, wpk_ga, 64, 1);
    nbfm_pack_w<<<dim3(144), T, 0, stream>>>(bbw, wpk_bb, 64, 1);
    nbfm_pack_w<<<dim3(144), T, 0, stream>>>(baw, wpk_ba, 64, 1);
    nbfm_pack_w<<<dim3(32),  T, 0, stream>>>(f1w, wpk_f1, 128, 0);

    // ---- add_style B ----
    nbfm_pack_concat<<<dim3(BS * 8192), T, 0, stream>>>(degraded, ref, actcat);
    nbfm_conv_wmma<128, 1><<<dim3(BS * 128), T, 0, stream>>>(actcat, wpk_sb, sbb,
                                                             (float*)nullptr, stylef, 1);
    nbfm_conv_wmma<64, 1><<<dim3(BS * 128), T, 0, stream>>>(stylef, wpk_gb, gbb,
                                                            gbuf, (_Float16*)nullptr, 0);
    nbfm_conv_wmma<64, 1><<<dim3(BS * 128), T, 0, stream>>>(stylef, wpk_bb, bbb,
                                                            bbuf, (_Float16*)nullptr, 0);
    nbfm_style_combine<<<dim3(BS * 4096), T, 0, stream>>>(ref, gbuf, bbuf,
                                                          sumR, sqR, sumD, sqD, ref2);

    // ---- attention ----
    nbfm_unfold_norm<<<dim3(BS * 512), T, 0, stream>>>(degraded, dpn, BS * 4096, 1, DL);
    nbfm_unfold_norm<<<dim3((BS * RL * RL + 7) / 8), T, 0, stream>>>(ref2, rpn,
                                                                     BS * RL * RL, 7, RL);
    nbfm_attn_matched<<<dim3(BS * 4096), T, 0, stream>>>(dpn, rpn, nbw, mcol);
    nbfm_fold<<<dim3(BS * 4096), T, 0, stream>>>(mcol, matched);

    // ---- add_style A ----
    nbfm_stats<<<dim3(BS * 64), T, 0, stream>>>(matched, sumM, sqM);
    nbfm_pack_concat<<<dim3(BS * 8192), T, 0, stream>>>(degraded, matched, actcat);
    nbfm_conv_wmma<128, 1><<<dim3(BS * 128), T, 0, stream>>>(actcat, wpk_sa, sab,
                                                             (float*)nullptr, stylef, 1);
    nbfm_conv_wmma<64, 1><<<dim3(BS * 128), T, 0, stream>>>(stylef, wpk_ga, gab,
                                                            gbuf, (_Float16*)nullptr, 0);
    nbfm_conv_wmma<64, 1><<<dim3(BS * 128), T, 0, stream>>>(stylef, wpk_ba, bab,
                                                            bbuf, (_Float16*)nullptr, 0);
    nbfm_style_combine<<<dim3(BS * 4096), T, 0, stream>>>(matched, gbuf, bbuf,
                                                          sumM, sqM, sumD, sqD, mstyled);

    // ---- fusion ----
    nbfm_pack_concat<<<dim3(BS * 8192), T, 0, stream>>>(degraded, mstyled, actcat);
    nbfm_conv_wmma<128, 0><<<dim3(BS * 128), T, 0, stream>>>(actcat, wpk_f1, f1b,
                                                             fus1out, (_Float16*)nullptr, 0);
    nbfm_dwconv<<<dim3(BS * 4096), T, 0, stream>>>(fus1out, f2w, f2b, (float*)d_out);
}
